// ATTNAligner_2095944040926
// MI455X (gfx1250) — matrix operand order repaired and checked by reading, not compiled, not run on hardware
//
#include <hip/hip_runtime.h>

// ---------------------------------------------------------------------------
// CDNA5 (gfx1250): bf16 WMMA GEMMs + persistent LSTM recurrence with
// TDM (tensor_load_to_lds) double-buffered gate-tile streaming.
// ---------------------------------------------------------------------------

#define B_SZ   16
#define T_SZ   1000
#define MEL_SZ 80
#define H_SZ   512
#define H2_SZ  256
#define G4_SZ  1024     // 4*H2
#define VOCAB  80
#define KW     5
#define ROWS   (T_SZ * B_SZ)   // 16000

#if defined(__has_builtin)
#  if __has_builtin(__builtin_amdgcn_tensor_load_to_lds)
#    define HAVE_TDM 1
#  endif
#  if __has_builtin(__builtin_amdgcn_sched_barrier)
#    define HAVE_SCHEDB 1
#  endif
#endif
#ifndef HAVE_TDM
#  define HAVE_TDM 0
#endif
#ifndef HAVE_SCHEDB
#  define HAVE_SCHEDB 0
#endif

#if HAVE_SCHEDB
#  define SCHED_FENCE() __builtin_amdgcn_sched_barrier(0)
#else
#  define SCHED_FENCE()
#endif

typedef __attribute__((ext_vector_type(16))) __bf16    v16bf;
typedef __attribute__((ext_vector_type(8)))  float     v8f;
typedef __attribute__((ext_vector_type(4)))  unsigned  v4u;
typedef __attribute__((ext_vector_type(8)))  int       v8i;
typedef __attribute__((ext_vector_type(4)))  int       v4i;

union Frag {
    v16bf   v;
    unsigned u[8];
};

__device__ inline unsigned short f2bf(float f) {
    unsigned u = __float_as_uint(f);
    unsigned r = u + 0x7FFFu + ((u >> 16) & 1u);   // round-to-nearest-even
    return (unsigned short)(r >> 16);
}
__device__ inline unsigned pack2(float a, float b) {
    return (unsigned)f2bf(a) | ((unsigned)f2bf(b) << 16);
}

__device__ inline v8f wmma_bf16(const Frag& a, const Frag& b, v8f c) {
    return __builtin_amdgcn_wmma_f32_16x16x32_bf16(
        false, a.v, false, b.v, (short)0, c, false, false);
}

// ---------------------------------------------------------------------------
// Prenet conv1 (MEL->H, causal K=5) + eval BN.  One block per (b,t).
// ---------------------------------------------------------------------------
__global__ __launch_bounds__(512) void conv1_bn(
    const float* __restrict__ mels, const float* __restrict__ w,
    const float* __restrict__ cb, const float* __restrict__ g,
    const float* __restrict__ bb, const float* __restrict__ bm,
    const float* __restrict__ bv, float* __restrict__ Y1)
{
    __shared__ float sX[MEL_SZ * KW];
    int blk = blockIdx.x;
    int b = blk / T_SZ, t = blk % T_SZ;
    for (int i = threadIdx.x; i < MEL_SZ * KW; i += 512) {
        int c = i / KW, dt = i % KW, tt = t - (KW - 1) + dt;
        float val = 0.f;
        if (tt >= 0) val = mels[(size_t)b * MEL_SZ * T_SZ + c * T_SZ + tt] * (1.f / 6.f) + 1.f;
        sX[i] = val;
    }
    __syncthreads();
    int h = threadIdx.x;
    const float* wr = w + (size_t)h * MEL_SZ * KW;
    float acc = 0.f;
    #pragma unroll 5
    for (int i = 0; i < MEL_SZ * KW; ++i) acc += sX[i] * wr[i];
    float sc = g[h] * rsqrtf(bv[h] + 1e-5f);
    Y1[(size_t)b * H_SZ * T_SZ + (size_t)h * T_SZ + t] = (acc + cb[h] - bm[h]) * sc + bb[h];
}

// conv2 (H->H, causal K=5) + BN + transpose to [T,B,H]
__global__ __launch_bounds__(512) void conv2_bn_tr(
    const float* __restrict__ Y1, const float* __restrict__ w,
    const float* __restrict__ cb, const float* __restrict__ g,
    const float* __restrict__ bb, const float* __restrict__ bm,
    const float* __restrict__ bv, float* __restrict__ XS)
{
    __shared__ float sX[H_SZ * KW];
    int blk = blockIdx.x;
    int b = blk / T_SZ, t = blk % T_SZ;
    for (int i = threadIdx.x; i < H_SZ * KW; i += 512) {
        int c = i / KW, dt = i % KW, tt = t - (KW - 1) + dt;
        sX[i] = (tt >= 0) ? Y1[(size_t)b * H_SZ * T_SZ + (size_t)c * T_SZ + tt] : 0.f;
    }
    __syncthreads();
    int h = threadIdx.x;
    const float* wr = w + (size_t)h * H_SZ * KW;
    float acc = 0.f;
    #pragma unroll 5
    for (int i = 0; i < H_SZ * KW; ++i) acc += sX[i] * wr[i];
    float sc = g[h] * rsqrtf(bv[h] + 1e-5f);
    XS[(size_t)t * B_SZ * H_SZ + (size_t)b * H_SZ + h] = (acc + cb[h] - bm[h]) * sc + bb[h];
}

// ---------------------------------------------------------------------------
// Generic bf16 WMMA GEMM:  C[M,N] = act(A[M,K]) @ W[N,K]^T * alpha + beta (+bias)
// Block tile 64x128, 8 waves (2x4), each wave 32x32 via 2x2 v_wmma 16x16x32.
// ---------------------------------------------------------------------------
#define LDSTR 21   // padded row stride in u32 (42 bf16) -> conflict-free ds_load_b32
template <int RELU>
__global__ __launch_bounds__(256) void wmma_gemm(
    const float* __restrict__ A, int lda,
    const float* __restrict__ W, int ldw,
    const float* __restrict__ bias,
    float* __restrict__ C, int ldc,
    int M, int N, int Kd, float alpha, float beta)
{
    __shared__ unsigned sA[64][LDSTR];
    __shared__ unsigned sW[128][LDSTR];
    int tid = threadIdx.x;
    int lane = tid & 31, wave = tid >> 5;
    int wm = wave >> 2, wn = wave & 3;
    int m0 = blockIdx.y * 64;
    int n0 = blockIdx.x * 128;

    v8f acc[2][2];
    #pragma unroll
    for (int i = 0; i < 2; ++i)
        #pragma unroll
        for (int j = 0; j < 2; ++j)
            #pragma unroll
            for (int e = 0; e < 8; ++e) acc[i][j][e] = 0.f;

    int half = lane >> 4, mr = lane & 15;
    int nK = Kd >> 5;
    for (int kb = 0; kb < nK; ++kb) {
        int k0 = kb << 5;
        {   // stage A tile 64x32 (f32 -> bf16), 8 floats / thread
            int r = tid >> 2, c0 = (tid & 3) << 3;
            int gm = m0 + r;
            float tmp[8];
            if (gm < M) {
                const float* p = A + (size_t)gm * lda + k0 + c0;
                #pragma unroll
                for (int i = 0; i < 8; ++i) {
                    float v = p[i];
                    tmp[i] = RELU ? fmaxf(v, 0.f) : v;
                }
                if (kb + 1 < nK) __builtin_prefetch(p + 32, 0, 3);
            } else {
                #pragma unroll
                for (int i = 0; i < 8; ++i) tmp[i] = 0.f;
            }
            #pragma unroll
            for (int i = 0; i < 4; ++i) sA[r][(c0 >> 1) + i] = pack2(tmp[2 * i], tmp[2 * i + 1]);
        }
        {   // stage W tile 128x32, 16 floats / thread
            int r = tid >> 1, c0 = (tid & 1) << 4;
            int gn = n0 + r;
            float tmp[16];
            if (gn < N) {
                const float* p = W + (size_t)gn * ldw + k0 + c0;
                #pragma unroll
                for (int i = 0; i < 16; ++i) tmp[i] = p[i];
                if (kb + 1 < nK) __builtin_prefetch(p + 32, 0, 3);
            } else {
                #pragma unroll
                for (int i = 0; i < 16; ++i) tmp[i] = 0.f;
            }
            #pragma unroll
            for (int i = 0; i < 8; ++i) sW[r][(c0 >> 1) + i] = pack2(tmp[2 * i], tmp[2 * i + 1]);
        }
        __syncthreads();

        Frag a[2], b[2];
        #pragma unroll
        for (int i = 0; i < 2; ++i) {       // A frags: ISA 16-bit 16x32 layout
            int row = wm * 32 + i * 16 + mr;
            #pragma unroll
            for (int v = 0; v < 8; ++v) {
                int k = (v < 4) ? (half * 8 + 2 * v) : (16 + half * 8 + 2 * (v - 4));
                a[i].u[v] = sA[row][k >> 1];
            }
        }
        #pragma unroll
        for (int j = 0; j < 2; ++j) {       // B frags: ISA 16-bit 32x16 layout
            int row = wn * 32 + j * 16 + mr;
            #pragma unroll
            for (int v = 0; v < 8; ++v) {
                int k = (v < 4) ? (half * 8 + 2 * v) : (16 + half * 8 + 2 * (v - 4));
                b[j].u[v] = sW[row][k >> 1];
            }
        }
        #pragma unroll
        for (int i = 0; i < 2; ++i)
            #pragma unroll
            for (int j = 0; j < 2; ++j)
                acc[i][j] = wmma_bf16(a[i], b[j], acc[i][j]);
        __syncthreads();
    }

    #pragma unroll
    for (int i = 0; i < 2; ++i)
        #pragma unroll
        for (int j = 0; j < 2; ++j)
            #pragma unroll
            for (int r = 0; r < 8; ++r) {
                int m = m0 + wm * 32 + i * 16 + half * 8 + r;
                int n = n0 + wn * 32 + j * 16 + mr;
                if (m < M && n < N) {
                    float v = acc[i][j][r] * alpha + beta;
                    if (bias) v += bias[n];
                    C[(size_t)m * ldc + n] = v;
                }
            }
}

// ---------------------------------------------------------------------------
// Pack Whh[2,1024,256] f32 into per-fragment bf16 layout:
//   Wp[dir][tile(64)][kstep(8)][lane(32)][v(8)]  (u32 = packed bf16 pair)
// ---------------------------------------------------------------------------
__global__ void pack_whh(const float* __restrict__ Whh, unsigned* __restrict__ Wp)
{
    int idx = blockIdx.x * 256 + threadIdx.x;
    if (idx >= 2 * 64 * 8 * 32 * 8) return;
    int v    = idx & 7;
    int lane = (idx >> 3) & 31;
    int ks   = (idx >> 8) & 7;
    int tile = (idx >> 11) & 63;
    int dir  = idx >> 17;
    int n = tile * 16 + (lane & 15);
    int k = ks * 32 + ((v < 4) ? ((lane >> 4) * 8 + 2 * v) : (16 + (lane >> 4) * 8 + 2 * (v - 4)));
    const float* w = Whh + (size_t)dir * G4_SZ * H2_SZ + (size_t)n * H2_SZ + k;
    Wp[idx] = pack2(w[0], w[1]);
}

// ---------------------------------------------------------------------------
// TDM: DMA one [16 x 1024] f32 gate tile (row stride 2048 floats) into LDS.
// D# per CDNA5 ISA: group0 = count/lds/global/type,
// group1 = data_size=4B, tile 1024x16, tensor_dim0_stride=2048.
// Issued by wave 0 only; tracked with TENSORcnt.  6-arg builtin form.
// ---------------------------------------------------------------------------
__device__ inline void tdm_load_gates(const float* gsrc, unsigned lds_off)
{
#if HAVE_TDM
    unsigned long long ga = (unsigned long long)(size_t)gsrc;
    v4u g0;
    g0[0] = 1u;                                           // count=1 (valid D#)
    g0[1] = lds_off;                                      // lds_addr (bytes)
    g0[2] = (unsigned)ga;                                 // global_addr[31:0]
    g0[3] = (unsigned)((ga >> 32) & 0x01FFFFFFu) | (2u << 30); // addr[56:32] | type=2
    v8i g1;
    g1[0] = (int)(2u << 16);                              // data_size = 4 bytes
    g1[1] = (int)(1024u << 16);                           // tensor_dim0 = 1024
    g1[2] = (int)(16u << 16);                             // tensor_dim1 = 16
    g1[3] = (int)(1024u << 16);                           // tile_dim0 = 1024
    g1[4] = 16;                                           // tile_dim1 = 16
    g1[5] = 2048;                                         // tensor_dim0_stride
    g1[6] = 0;
    g1[7] = 0;
    v4i gz4;
    gz4[0] = 0; gz4[1] = 0; gz4[2] = 0; gz4[3] = 0;
    v8i gz8;
    #pragma unroll
    for (int i = 0; i < 8; ++i) gz8[i] = 0;
    __builtin_amdgcn_tensor_load_to_lds(g0, g1, gz4, gz4, gz8, 0);
#else
    (void)gsrc; (void)lds_off;
#endif
}

// ---------------------------------------------------------------------------
// Persistent bidirectional LSTM recurrence.  grid = 2 (dir), block = 512 (16 waves).
// Gates = Gin[t] (bias folded, TDM-streamed into LDS) + h_prev @ Whh^T via WMMA.
// h kept in LDS as bf16; cell state in registers (8 elems / thread).
// Weight fragments stream from L2 as global_load_b128 each step (opaque scalar
// offset defeats LICM without breaking addrspace inference); sched_barrier
// fences keep the next tile's 16 loads issued ahead of the current tile's
// 8-WMMA chain so waits are partial (loadcnt<=16), not per-fragment stalls.
// ---------------------------------------------------------------------------
__global__ __launch_bounds__(512) void lstm_recur(
    const float* __restrict__ Gin,     // [T, B, 2048]; this dir at offset dir*1024
    const unsigned* __restrict__ Wp,   // packed Whh fragments
    float* __restrict__ Xout)          // [T, B, 512]; dir slice at dir*256
{
    __shared__ float    gload[2][B_SZ][G4_SZ];   // double-buffered gates (f32)
    __shared__ unsigned hbuf[B_SZ][H2_SZ / 2];   // h, packed bf16 pairs

    int dir  = blockIdx.x;
    int tid  = threadIdx.x;
    int lane = tid & 31, wave = tid >> 5;
    int half = lane >> 4, mr = lane & 15;

    for (int i = tid; i < B_SZ * (H2_SZ / 2); i += 512) ((unsigned*)hbuf)[i] = 0u;

    float cst[8];
    #pragma unroll
    for (int i = 0; i < 8; ++i) cst[i] = 0.f;
    int base = tid * 8;
    int ebat = base >> 8;     // batch index for elementwise phase
    int ej   = base & 255;    // first gate-slot index
    const unsigned* WpD = Wp + (size_t)dir * 64 * 8 * 32 * 8;

    unsigned ldsbase = (unsigned)(size_t)&gload[0][0][0];

#if HAVE_TDM
    if (wave == 0) {   // prime the pipeline: tiles for steps 0 and 1
        int te0 = dir ? (T_SZ - 1) : 0;
        int te1 = dir ? (T_SZ - 2) : 1;
        tdm_load_gates(Gin + (size_t)te0 * B_SZ * 2048 + dir * G4_SZ, ldsbase);
        tdm_load_gates(Gin + (size_t)te1 * B_SZ * 2048 + dir * G4_SZ, ldsbase + 65536u);
    }
#endif
    __syncthreads();

    for (int t = 0; t < T_SZ; ++t) {
        int teff = dir ? (T_SZ - 1 - t) : t;
        int cur = t & 1;
        const float* g0 = Gin + (size_t)teff * B_SZ * 2048 + dir * G4_SZ;

#if HAVE_TDM
        if (wave == 0) __builtin_amdgcn_s_wait_tensorcnt((short)1); // tile t landed
        __syncthreads();                                            // publish to all waves
#else
        for (int i = tid; i < B_SZ * G4_SZ; i += 512)
            gload[cur][0][i] = g0[(size_t)(i >> 10) * 2048 + (i & 1023)];
        __syncthreads();
#endif
        (void)g0;

        // ---- matrix phase ----------------------------------------------
        // A fragments (h_prev) for all 8 k-steps, shared by the 4 N-tiles.
        Frag afr[8];
        #pragma unroll
        for (int ks = 0; ks < 8; ++ks) {
            #pragma unroll
            for (int v = 0; v < 8; ++v) {
                int k = ks * 32 + ((v < 4) ? (half * 8 + 2 * v)
                                           : (16 + half * 8 + 2 * (v - 4)));
                afr[ks].u[v] = hbuf[mr][k >> 1];
            }
        }

        // Opaque scalar offset (redefined every iteration by asm volatile):
        // stops LICM from hoisting 1000 steps' worth of loop-invariant weight
        // loads into (spilled) registers, while the pointer base keeps its
        // global address space -> global_load_b128, not flat_load.
        unsigned wofs = 0;
        asm volatile("" : "+s"(wofs));
        const unsigned* WpT = WpD + wofs;

        Frag bB[2][8];   // double-buffered B fragments (2 x 64 VGPRs)
        {
            const unsigned* p0 = WpT + (size_t)(wave * 4) * 8 * 256 + lane * 8;
            #pragma unroll
            for (int ks = 0; ks < 8; ++ks)
                #pragma unroll
                for (int v = 0; v < 8; ++v) bB[0][ks].u[v] = p0[ks * 256 + v];
        }
        #pragma unroll
        for (int tt = 0; tt < 4; ++tt) {
            int tile = wave * 4 + tt;
            int bufc = tt & 1;
            if (tt < 3) {   // prefetch next tile's fragments during this tile's WMMAs
                const unsigned* pn = WpT + (size_t)(tile + 1) * 8 * 256 + lane * 8;
                #pragma unroll
                for (int ks = 0; ks < 8; ++ks)
                    #pragma unroll
                    for (int v = 0; v < 8; ++v) bB[bufc ^ 1][ks].u[v] = pn[ks * 256 + v];
            }
            SCHED_FENCE();   // prefetch group must stay ahead of the math below
            v8f acc;
            #pragma unroll
            for (int r = 0; r < 8; ++r)
                acc[r] = gload[cur][half * 8 + r][tile * 16 + mr];
            #pragma unroll
            for (int ks = 0; ks < 8; ++ks)
                acc = wmma_bf16(afr[ks], bB[bufc][ks], acc);
            #pragma unroll
            for (int r = 0; r < 8; ++r) gload[cur][half * 8 + r][tile * 16 + mr] = acc[r];
            SCHED_FENCE();   // and the math must not sink into the next prefetch
        }
        __syncthreads();

        // ---- elementwise LSTM cell (gate order i, f, g, o) -------------
        float hv[8];
        #pragma unroll
        for (int i = 0; i < 8; ++i) {
            int j = ej + i;
            float gi = gload[cur][ebat][j];
            float gf = gload[cur][ebat][256 + j];
            float gg = gload[cur][ebat][512 + j];
            float go = gload[cur][ebat][768 + j];
            float si = 1.f / (1.f + __expf(-gi));
            float sf = 1.f / (1.f + __expf(-gf));
            float so = 1.f / (1.f + __expf(-go));
            float c = sf * cst[i] + si * tanhf(gg);
            cst[i] = c;
            hv[i] = so * tanhf(c);
        }
        #pragma unroll
        for (int i = 0; i < 4; ++i)
            hbuf[ebat][(ej >> 1) + i] = pack2(hv[2 * i], hv[2 * i + 1]);
        float* xo = Xout + (size_t)teff * B_SZ * H_SZ + (size_t)ebat * H_SZ + dir * H2_SZ + ej;
        #pragma unroll
        for (int i = 0; i < 8; ++i) xo[i] = hv[i];
        __syncthreads();   // everyone done with buffer `cur`

#if HAVE_TDM
        if (wave == 0 && t + 2 < T_SZ) {   // refill `cur` with tile for step t+2
            int tn = dir ? (T_SZ - 1 - (t + 2)) : (t + 2);
            tdm_load_gates(Gin + (size_t)tn * B_SZ * 2048 + dir * G4_SZ,
                           ldsbase + (unsigned)cur * 65536u);
        }
#endif
    }
}

// ---------------------------------------------------------------------------
// Row LayerNorm over last dim D=512 (in place).  1 block / row, 256 threads.
// ---------------------------------------------------------------------------
__global__ __launch_bounds__(256) void layernorm_rows(
    float* __restrict__ X, const float* __restrict__ g, const float* __restrict__ b)
{
    __shared__ float red[256];
    int row = blockIdx.x;
    float* x = X + (size_t)row * H_SZ;
    int i0 = threadIdx.x * 2;
    float v0 = x[i0], v1 = x[i0 + 1];
    red[threadIdx.x] = v0 + v1;
    __syncthreads();
    for (int o = 128; o > 0; o >>= 1) {
        if (threadIdx.x < o) red[threadIdx.x] += red[threadIdx.x + o];
        __syncthreads();
    }
    float mu = red[0] * (1.f / H_SZ);
    __syncthreads();
    float d0 = v0 - mu, d1 = v1 - mu;
    red[threadIdx.x] = d0 * d0 + d1 * d1;
    __syncthreads();
    for (int o = 128; o > 0; o >>= 1) {
        if (threadIdx.x < o) red[threadIdx.x] += red[threadIdx.x + o];
        __syncthreads();
    }
    float inv = rsqrtf(red[0] * (1.f / H_SZ) + 1e-5f);
    x[i0]     = d0 * inv * g[i0]     + b[i0];
    x[i0 + 1] = d1 * inv * g[i0 + 1] + b[i0 + 1];
}

// concat [rows,512]+[rows,512] -> [rows,1024]
__global__ void concat_rows(const float* __restrict__ A, const float* __restrict__ Bc,
                            float* __restrict__ C, int rows)
{
    int idx = blockIdx.x * 256 + threadIdx.x;
    if (idx >= rows * 1024) return;
    int r = idx >> 10, c = idx & 1023;
    C[idx] = (c < 512) ? A[(size_t)r * 512 + c] : Bc[(size_t)r * 512 + c - 512];
}

// masked log_softmax over VOCAB=80, rows ordered [T,B]
__global__ __launch_bounds__(128) void ctc_logsoftmax(
    const float* __restrict__ logits, const int* __restrict__ mel_len,
    float* __restrict__ out)
{
    __shared__ float red[128];
    int row = blockIdx.x;
    int tid = threadIdx.x;
    const float* x = logits + (size_t)row * VOCAB;
    float v = (tid < VOCAB) ? x[tid] : -1e30f;
    red[tid] = v;
    __syncthreads();
    for (int o = 64; o > 0; o >>= 1) {
        if (tid < o) red[tid] = fmaxf(red[tid], red[tid + o]);
        __syncthreads();
    }
    float mx = red[0];
    __syncthreads();
    float e = (tid < VOCAB) ? __expf(v - mx) : 0.f;
    red[tid] = e;
    __syncthreads();
    for (int o = 64; o > 0; o >>= 1) {
        if (tid < o) red[tid] += red[tid + o];
        __syncthreads();
    }
    float lse = __logf(red[0]);
    int t = row / B_SZ, b = row % B_SZ;
    float mask = (t < mel_len[b]) ? 1.f : 0.f;
    if (tid < VOCAB) out[(size_t)row * VOCAB + tid] = (v - mx - lse) * mask;
}

// ---------------------------------------------------------------------------
extern "C" void kernel_launch(void* const* d_in, const int* in_sizes, int n_in,
                              void* d_out, int out_size, void* d_ws, size_t ws_size,
                              hipStream_t stream)
{
    const float* mels = (const float*)d_in[0];
    const float* c1w = (const float*)d_in[1];  const float* c1b = (const float*)d_in[2];
    const float* bn1g = (const float*)d_in[3]; const float* bn1b = (const float*)d_in[4];
    const float* bn1m = (const float*)d_in[5]; const float* bn1v = (const float*)d_in[6];
    const float* c2w = (const float*)d_in[7];  const float* c2b = (const float*)d_in[8];
    const float* bn2g = (const float*)d_in[9]; const float* bn2b = (const float*)d_in[10];
    const float* bn2m = (const float*)d_in[11];const float* bn2v = (const float*)d_in[12];
    const float* eWih = (const float*)d_in[13];const float* eWhh = (const float*)d_in[14];
    const float* eb   = (const float*)d_in[15];
    const float* elng = (const float*)d_in[16];const float* elnb = (const float*)d_in[17];
    const float* cWih = (const float*)d_in[18];const float* cWhh = (const float*)d_in[19];
    const float* cb   = (const float*)d_in[20];
    const float* ctcW = (const float*)d_in[21];
    const float* btlW = (const float*)d_in[22];
    const float* lng  = (const float*)d_in[23];const float* lnb = (const float*)d_in[24];
    const float* dWih = (const float*)d_in[25];const float* dWhh = (const float*)d_in[26];
    const float* db   = (const float*)d_in[27];
    const float* melW = (const float*)d_in[28];
    const int*   mel_lengths = (const int*)d_in[29];
    float* out = (float*)d_out;

    // workspace layout (floats)
    float* ws  = (float*)d_ws;
    float* Y1  = ws;                        // [B,H,T]  8,192,000 (reused as encoder h)
    float* XS  = Y1 + 8192000;              // [T,B,H]
    float* T0  = XS + 8192000;
    float* T1v = T0 + 8192000;
    float* CX  = T1v + 8192000;
    float* GIN = CX + 8192000;              // [T,B,2048] (also concat buffer)
    float* LG  = GIN + 32768000;            // logits [T,B,80]
    unsigned* WP = (unsigned*)(LG + 1280000); // packed Whh: 262144 u32

    const int lOffIh = 2 * G4_SZ * H_SZ;    // per-layer stride in Wih
    const int lOffHh = 2 * G4_SZ * H2_SZ;
    const int lOffB  = 2 * G4_SZ;

    auto do_layer = [&](const float* Wih_l, const float* Whh_l, const float* b_l,
                        const float* xin, float* xout) {
        pack_whh<<<(2 * 64 * 8 * 32 * 8 + 255) / 256, 256, 0, stream>>>(Whh_l, WP);
        dim3 gg((2048 + 127) / 128, (ROWS + 63) / 64);
        wmma_gemm<0><<<gg, 256, 0, stream>>>(xin, H_SZ, Wih_l, H_SZ, b_l,
                                             GIN, 2048, ROWS, 2048, H_SZ, 1.f, 0.f);
        lstm_recur<<<2, 512, 0, stream>>>(GIN, WP, xout);
    };

    // prenet
    conv1_bn<<<B_SZ * T_SZ, 512, 0, stream>>>(mels, c1w, c1b, bn1g, bn1b, bn1m, bn1v, Y1);
    conv2_bn_tr<<<B_SZ * T_SZ, 512, 0, stream>>>(Y1, c2w, c2b, bn2g, bn2b, bn2m, bn2v, XS);

    // encoder BiLSTM (2 layers) + LN  -> Y1 holds decoder_input h
    do_layer(eWih, eWhh, eb, XS, T0);
    do_layer(eWih + lOffIh, eWhh + lOffHh, eb + lOffB, T0, Y1);
    layernorm_rows<<<ROWS, 256, 0, stream>>>(Y1, elng, elnb);

    // CTC BiLSTM (2 layers) -> CX
    do_layer(cWih, cWhh, cb, Y1, T0);
    do_layer(cWih + lOffIh, cWhh + lOffHh, cb + lOffB, T0, CX);

    // logits = relu(CX) @ ctcW^T ; masked log_softmax -> second half of d_out
    wmma_gemm<1><<<dim3(1, (ROWS + 63) / 64), 256, 0, stream>>>(
        CX, H_SZ, ctcW, H_SZ, nullptr, LG, VOCAB, ROWS, VOCAB, H_SZ, 1.f, 0.f);
    ctc_logsoftmax<<<ROWS, 128, 0, stream>>>(LG, mel_lengths, out + (size_t)ROWS * VOCAB);

    // bottleneck: LN( cat(h, cx) @ btlW^T ) -> T0
    concat_rows<<<(ROWS * 1024 + 255) / 256, 256, 0, stream>>>(Y1, CX, GIN, ROWS);
    wmma_gemm<0><<<dim3(4, (ROWS + 63) / 64), 256, 0, stream>>>(
        GIN, 1024, btlW, 1024, nullptr, T0, H_SZ, ROWS, H_SZ, 1024, 1.f, 0.f);
    layernorm_rows<<<ROWS, 256, 0, stream>>>(T0, lng, lnb);

    // mel decoder BiLSTM (2 layers) -> T0
    do_layer(dWih, dWhh, db, T0, T1v);
    do_layer(dWih + lOffIh, dWhh + lOffHh, db + lOffB, T1v, T0);

    // mel = denorm( mx @ melW^T ) = 6*x - 6  -> first half of d_out
    wmma_gemm<0><<<dim3(1, (ROWS + 63) / 64), 256, 0, stream>>>(
        T0, H_SZ, melW, H_SZ, nullptr, out, VOCAB, ROWS, MEL_SZ, H_SZ, 6.f, -6.f);

    (void)in_sizes; (void)n_in; (void)out_size; (void)ws_size;
}
